// MultiModalFusionModule_8211977470440
// MI455X (gfx1250) — compile-verified, matrix operand order checked by reading, not run on hardware
//
#include <hip/hip_runtime.h>
#include <math.h>

#define NN 50000
#define DD 128
#define EE 800000
#define THRB 0.3f
#define LN_EPS 1e-5f

typedef __attribute__((ext_vector_type(16))) __bf16 v16bf;
typedef __attribute__((ext_vector_type(8)))  float v8f;

// ---------------------------------------------------------------------------
// 1. Boundary MLP per node: 128 -> 64 (relu) -> 32 (relu) -> 1 (sigmoid)
// one thread per node; weights staged in LDS (broadcast reads, no conflicts)
// ---------------------------------------------------------------------------
__global__ void boundary_kernel(const float* __restrict__ x,
                                const float* __restrict__ W1, const float* __restrict__ b1,
                                const float* __restrict__ W2, const float* __restrict__ b2,
                                const float* __restrict__ W3, const float* __restrict__ b3,
                                float* __restrict__ bout)
{
    __shared__ float sW1[DD * 64];   // 32 KB
    __shared__ float sW2[64 * 32];   // 8 KB
    __shared__ float sW3[32];
    __shared__ float sb1[64], sb2[32], sb3s[1];
    const int t   = blockIdx.y;
    const int tid = threadIdx.x;
    for (int i = tid; i < DD * 64; i += blockDim.x) sW1[i] = W1[(size_t)t * DD * 64 + i];
    for (int i = tid; i < 64 * 32; i += blockDim.x) sW2[i] = W2[(size_t)t * 64 * 32 + i];
    if (tid < 64) sb1[tid] = b1[t * 64 + tid];
    if (tid < 32) { sW3[tid] = W3[t * 32 + tid]; sb2[tid] = b2[t * 32 + tid]; }
    if (tid == 0) sb3s[0] = b3[t];
    __syncthreads();

    const int n = blockIdx.x * blockDim.x + tid;
    if (n >= NN) return;

    float xr[DD];
    const float* xp = x + ((size_t)t * NN + n) * DD;
    for (int k = 0; k < DD; ++k) xr[k] = xp[k];

    float h2[32];
    for (int j2 = 0; j2 < 32; ++j2) h2[j2] = sb2[j2];
    for (int j = 0; j < 64; ++j) {
        float a = sb1[j];
        for (int k = 0; k < DD; ++k) a = fmaf(xr[k], sW1[k * 64 + j], a);
        a = fmaxf(a, 0.f);
        for (int j2 = 0; j2 < 32; ++j2) h2[j2] = fmaf(a, sW2[j * 32 + j2], h2[j2]);
    }
    float o = sb3s[0];
    for (int j2 = 0; j2 < 32; ++j2) o = fmaf(fmaxf(h2[j2], 0.f), sW3[j2], o);
    bout[(size_t)t * NN + n] = 1.f / (1.f + expf(-o));
}

// ---------------------------------------------------------------------------
// 2. WMMA GEMM: Out[nrows,128] = act(A[nrows,128] @ W[128,128] + bias)
// bf16 inputs, f32 accumulation, V_WMMA_F32_16X16X32_BF16.
// 256 threads = 8 waves; wave w owns rows [m0+w*16, +16) x all 128 cols.
// W is staged in LDS *pre-swizzled into B-fragment layout*: one 32-byte
// v16bf per (frag = kc*8+nt, lane), so the hot loop does 2x ds_load_b128
// per fragment instead of 16x ds_load_u16.
// ---------------------------------------------------------------------------
__global__ void gemm_wmma_kernel(const float* __restrict__ A,
                                 const float* __restrict__ W,
                                 const float* __restrict__ bias,
                                 float* __restrict__ Out,
                                 int nrows, int do_relu)
{
    __shared__ v16bf sWf[32 * 32];   // [frag(kc*8+nt)][lane], 32 KB
    __shared__ float sBias[DD];
    const int tid = threadIdx.x;

    // Stage W into swizzled fragment layout. Each slot = 16 bf16 halves of
    // one lane's B-fragment, written as a single 32-byte LDS store pair.
    for (int slot = tid; slot < 32 * 32; slot += 256) {
        const int lane_s = slot & 31;
        const int frag   = slot >> 5;        // kc*8 + nt
        const int kc_s   = frag >> 3;
        const int nt_s   = frag & 7;
        const int col    = nt_s * 16 + (lane_s & 15);
        const int kb     = kc_s * 32 + (lane_s >> 4) * 8;
        v16bf f;
#pragma unroll
        for (int h = 0; h < 8; ++h) {
            f[h]     = (__bf16)W[(kb + h) * DD + col];
            f[h + 8] = (__bf16)W[(kb + 16 + h) * DD + col];
        }
        sWf[slot] = f;
    }
    if (tid < DD) sBias[tid] = bias[tid];
    __syncthreads();

    const int wave = tid >> 5;
    const int lane = tid & 31;
    const int lm   = lane & 15;   // M for A-frag, N for B-frag / C,D column
    const int lh   = lane >> 4;   // selects K-base (0 or 8) for A/B halves
    const int m0   = blockIdx.x * 128 + wave * 16;
    const int arow = m0 + lm;
    const bool rok = arow < nrows;
    const int  crow = rok ? arow : 0;     // clamp: keep loads in-bounds,
    const float rs  = rok ? 1.f : 0.f;    // zero the contribution instead
    const float* ap = A + (size_t)crow * DD;

    v8f zero = {0.f, 0.f, 0.f, 0.f, 0.f, 0.f, 0.f, 0.f};
    v8f acc[8];
#pragma unroll
    for (int nt = 0; nt < 8; ++nt) acc[nt] = zero;

#pragma unroll
    for (int kc = 0; kc < 4; ++kc) {           // K = 128 in chunks of 32
        const int kb = kc * 32 + lh * 8;
        const float4 a0 = *(const float4*)(ap + kb);
        const float4 a1 = *(const float4*)(ap + kb + 4);
        const float4 a2 = *(const float4*)(ap + kb + 16);
        const float4 a3 = *(const float4*)(ap + kb + 20);
        v16bf afrag;
        afrag[0]  = (__bf16)(a0.x * rs); afrag[1]  = (__bf16)(a0.y * rs);
        afrag[2]  = (__bf16)(a0.z * rs); afrag[3]  = (__bf16)(a0.w * rs);
        afrag[4]  = (__bf16)(a1.x * rs); afrag[5]  = (__bf16)(a1.y * rs);
        afrag[6]  = (__bf16)(a1.z * rs); afrag[7]  = (__bf16)(a1.w * rs);
        afrag[8]  = (__bf16)(a2.x * rs); afrag[9]  = (__bf16)(a2.y * rs);
        afrag[10] = (__bf16)(a2.z * rs); afrag[11] = (__bf16)(a2.w * rs);
        afrag[12] = (__bf16)(a3.x * rs); afrag[13] = (__bf16)(a3.y * rs);
        afrag[14] = (__bf16)(a3.z * rs); afrag[15] = (__bf16)(a3.w * rs);
#pragma unroll
        for (int nt = 0; nt < 8; ++nt) {
            const v16bf bfrag = sWf[(kc * 8 + nt) * 32 + lane];
            acc[nt] = __builtin_amdgcn_wmma_f32_16x16x32_bf16(
                false, afrag, false, bfrag, (short)0, acc[nt], false, false);
        }
    }

#pragma unroll
    for (int nt = 0; nt < 8; ++nt) {
        const int col = nt * 16 + lm;
        const float bv = sBias[col];
#pragma unroll
        for (int v = 0; v < 8; ++v) {
            const int m = m0 + lh * 8 + v;     // C/D layout: VGPR v -> M = v + 8*lh
            if (m < nrows) {
                float r = acc[nt][v] + bv;
                if (do_relu) r = fmaxf(r, 0.f);
                Out[(size_t)m * DD + col] = r;
            }
        }
    }
}

// ---------------------------------------------------------------------------
// 3. Per-edge boundary-weight MLP: [sb,db] -> 32 (relu) -> 16 (relu) -> sigmoid
//    coef = ea * (enhanced) w
// ---------------------------------------------------------------------------
__global__ void edgeweight_kernel(const float* __restrict__ bsc,
                                  const int* __restrict__ ei,
                                  const float* __restrict__ ea,
                                  const float* __restrict__ W1, const float* __restrict__ b1,
                                  const float* __restrict__ W2, const float* __restrict__ b2,
                                  const float* __restrict__ W3, const float* __restrict__ b3,
                                  float* __restrict__ coef)
{
    __shared__ float sW1[64], sb1[32], sW2[32 * 16], sb2[16], sW3[16], sb3s[1];
    const int e = blockIdx.y;
    const int tid = threadIdx.x;
    if (tid < 64) sW1[tid] = W1[e * 64 + tid];
    if (tid < 32) sb1[tid] = b1[e * 32 + tid];
    for (int i = tid; i < 512; i += blockDim.x) sW2[i] = W2[e * 512 + i];
    if (tid < 16) { sb2[tid] = b2[e * 16 + tid]; sW3[tid] = W3[e * 16 + tid]; }
    if (tid == 0) sb3s[0] = b3[e];
    __syncthreads();

    const int k = blockIdx.x * blockDim.x + tid;
    if (k >= EE) return;
    const int src = ei[(size_t)e * 2 * EE + k];
    const int dst = ei[(size_t)e * 2 * EE + EE + k];
    const float sb = bsc[(size_t)e * NN + src];
    const float db = bsc[(size_t)(1 - e) * NN + dst];

    float h2[16];
    for (int j2 = 0; j2 < 16; ++j2) h2[j2] = sb2[j2];
    for (int j = 0; j < 32; ++j) {
        float a = fmaf(db, sW1[32 + j], fmaf(sb, sW1[j], sb1[j]));
        a = fmaxf(a, 0.f);
        for (int j2 = 0; j2 < 16; ++j2) h2[j2] = fmaf(a, sW2[j * 16 + j2], h2[j2]);
    }
    float o = sb3s[0];
    for (int j2 = 0; j2 < 16; ++j2) o = fmaf(fmaxf(h2[j2], 0.f), sW3[j2], o);
    float w = 1.f / (1.f + expf(-o));
    if (sb > THRB || db > THRB) w *= 2.f;
    coef[(size_t)e * EE + k] = ea[(size_t)e * EE + k] * w;
}

// ---------------------------------------------------------------------------
// 4. Zero aggregation buffer (deterministic; graph-capture safe)
// ---------------------------------------------------------------------------
__global__ void zero_kernel(float* __restrict__ p, long long n)
{
    long long i = (long long)blockIdx.x * blockDim.x + threadIdx.x;
    if (i < n) p[i] = 0.f;
}

// ---------------------------------------------------------------------------
// 5. Gather t[src], scale by coef, atomic scatter-add into aggr[dst].
// 32 threads per edge, float4 per thread, f32 atomics resolved in L2.
// ---------------------------------------------------------------------------
__global__ void scatter_kernel(const float* __restrict__ tmsg,
                               const float* __restrict__ coef,
                               const int* __restrict__ ei,
                               float* __restrict__ aggr)
{
    long long id = (long long)blockIdx.x * blockDim.x + threadIdx.x;
    if (id >= (long long)2 * EE * 32) return;
    const int g = (int)(id & 31);
    long long rem = id >> 5;
    const int k = (int)(rem % EE);
    const int e = (int)(rem / EE);
    const int src = ei[(size_t)e * 2 * EE + k];
    const int dst = ei[(size_t)e * 2 * EE + EE + k];
    const float c = coef[(size_t)e * EE + k];
    const float4 v = *(const float4*)(tmsg + ((size_t)e * NN + src) * DD + g * 4);
    float* ap = aggr + ((size_t)e * NN + dst) * DD + g * 4;
    atomicAdd(ap + 0, v.x * c);
    atomicAdd(ap + 1, v.y * c);
    atomicAdd(ap + 2, v.z * c);
    atomicAdd(ap + 3, v.w * c);
}

// ---------------------------------------------------------------------------
// 6. updated[tt] = aggr[1-tt] + x[tt]
// ---------------------------------------------------------------------------
__global__ void residual_kernel(const float* __restrict__ aggr,
                                const float* __restrict__ x,
                                float* __restrict__ upd)
{
    long long i = (long long)blockIdx.x * blockDim.x + threadIdx.x;
    const long long half = (long long)NN * DD;
    if (i >= 2 * half) return;
    const int tt = (int)(i >= half);
    const long long r = i - (long long)tt * half;
    upd[i] = aggr[(size_t)(1 - tt) * half + r] + x[i];
}

// ---------------------------------------------------------------------------
// 8. LayerNorm + affine + ReLU over last dim (128). One 128-thread block/row.
// ---------------------------------------------------------------------------
__global__ void ln_kernel(const float* __restrict__ z,
                          const float* __restrict__ lng,
                          const float* __restrict__ lnb,
                          float* __restrict__ out)
{
    __shared__ float red[DD];
    const int row = blockIdx.x;
    const int tt  = row / NN;
    const int tid = threadIdx.x;
    const float v = z[(size_t)row * DD + tid];

    red[tid] = v; __syncthreads();
    for (int s = 64; s > 0; s >>= 1) { if (tid < s) red[tid] += red[tid + s]; __syncthreads(); }
    const float mu = red[0] / DD; __syncthreads();

    const float d = v - mu;
    red[tid] = d * d; __syncthreads();
    for (int s = 64; s > 0; s >>= 1) { if (tid < s) red[tid] += red[tid + s]; __syncthreads(); }
    const float var = red[0] / DD;

    const float zn = d * rsqrtf(var + LN_EPS);
    const float r = fmaf(zn, lng[tt * DD + tid], lnb[tt * DD + tid]);
    out[(size_t)row * DD + tid] = fmaxf(r, 0.f);
}

// ---------------------------------------------------------------------------
extern "C" void kernel_launch(void* const* d_in, const int* in_sizes, int n_in,
                              void* d_out, int out_size, void* d_ws, size_t ws_size,
                              hipStream_t stream)
{
    (void)in_sizes; (void)n_in; (void)out_size; (void)ws_size;
    const float* x    = (const float*)d_in[0];
    const int*   ei   = (const int*)d_in[1];
    const float* ea   = (const float*)d_in[2];
    const float* bdW1 = (const float*)d_in[3];
    const float* bdb1 = (const float*)d_in[4];
    const float* bdW2 = (const float*)d_in[5];
    const float* bdb2 = (const float*)d_in[6];
    const float* bdW3 = (const float*)d_in[7];
    const float* bdb3 = (const float*)d_in[8];
    const float* etW1 = (const float*)d_in[9];
    const float* etb1 = (const float*)d_in[10];
    const float* etW2 = (const float*)d_in[11];
    const float* etb2 = (const float*)d_in[12];
    const float* bwW1 = (const float*)d_in[13];
    const float* bwb1 = (const float*)d_in[14];
    const float* bwW2 = (const float*)d_in[15];
    const float* bwb2 = (const float*)d_in[16];
    const float* bwW3 = (const float*)d_in[17];
    const float* bwb3 = (const float*)d_in[18];
    const float* nuW  = (const float*)d_in[19];
    const float* nub  = (const float*)d_in[20];
    const float* lng  = (const float*)d_in[21];
    const float* lnb  = (const float*)d_in[22];

    // Workspace partition (floats): b | coef | h | t(=z) | aggr  (~160 MB)
    float* ws   = (float*)d_ws;
    float* b_sc = ws;
    float* coef = b_sc + 2LL * NN;
    float* hbuf = coef + 2LL * EE;
    float* tbuf = hbuf + 2LL * NN * DD;
    float* aggr = tbuf + 2LL * NN * DD;

    // 1. boundary scores
    dim3 bgrid((NN + 127) / 128, 2);
    boundary_kernel<<<bgrid, 128, 0, stream>>>(x, bdW1, bdb1, bdW2, bdb2, bdW3, bdb3, b_sc);

    // 2. edge-type node transforms: t = relu(x@W1+b1)@W2+b2  (WMMA)
    const int gblocks = (NN + 127) / 128;
    for (int e = 0; e < 2; ++e) {
        gemm_wmma_kernel<<<gblocks, 256, 0, stream>>>(
            x + (size_t)e * NN * DD, etW1 + (size_t)e * DD * DD, etb1 + (size_t)e * DD,
            hbuf + (size_t)e * NN * DD, NN, 1);
        gemm_wmma_kernel<<<gblocks, 256, 0, stream>>>(
            hbuf + (size_t)e * NN * DD, etW2 + (size_t)e * DD * DD, etb2 + (size_t)e * DD,
            tbuf + (size_t)e * NN * DD, NN, 0);
    }

    // 3. per-edge weights
    dim3 egrid((EE + 255) / 256, 2);
    edgeweight_kernel<<<egrid, 256, 0, stream>>>(b_sc, ei, ea,
                                                 bwW1, bwb1, bwW2, bwb2, bwW3, bwb3, coef);

    // 4. zero aggr
    const long long na = 2LL * NN * DD;
    zero_kernel<<<(int)((na + 255) / 256), 256, 0, stream>>>(aggr, na);

    // 5. gather-scale-scatter (roofline-dominant phase)
    const long long nsc = 2LL * EE * 32;
    scatter_kernel<<<(int)((nsc + 255) / 256), 256, 0, stream>>>(tbuf, coef, ei, aggr);

    // 6. residual: upd[tt] = aggr[1-tt] + x[tt]   (into hbuf)
    residual_kernel<<<(int)((na + 255) / 256), 256, 0, stream>>>(aggr, x, hbuf);

    // 7. node update GEMM (WMMA) -> z (reuse tbuf)
    for (int tt = 0; tt < 2; ++tt) {
        gemm_wmma_kernel<<<gblocks, 256, 0, stream>>>(
            hbuf + (size_t)tt * NN * DD, nuW + (size_t)tt * DD * DD, nub + (size_t)tt * DD,
            tbuf + (size_t)tt * NN * DD, NN, 0);
    }

    // 8. LayerNorm + ReLU -> output
    ln_kernel<<<2 * NN, DD, 0, stream>>>(tbuf, lng, lnb, (float*)d_out);
}
